// LSTMEDModule_53042846105760
// MI455X (gfx1250) — matrix operand order repaired and checked by reading, not compiled, not run on hardware
//
#include <hip/hip_runtime.h>
#include <hip/hip_bf16.h>

// ---- problem dims (fixed by the reference) ----
#define BB 128
#define TT 256
#define FF 64
#define HH 1024
#define G4H (4 * HH)

typedef _Float16 f16;
typedef __attribute__((ext_vector_type(16))) _Float16 v16h;
typedef __attribute__((ext_vector_type(8)))  float    v8f;
typedef __attribute__((ext_vector_type(4)))  int      v4i;

// ---------------- async global->LDS staging (CDNA5), with safe fallback ----------------
#if defined(__has_builtin)
#if __has_builtin(__builtin_amdgcn_global_load_async_to_lds_b128)
#define HAVE_ASYNC_LDS 1
#endif
#endif
#ifndef HAVE_ASYNC_LDS
#define HAVE_ASYNC_LDS 0
#endif

__device__ __forceinline__ void stage16B(const f16* g, f16* l) {
#if HAVE_ASYNC_LDS
  // signature (probe-confirmed): (v4i AS1*, v4i AS3*, Imm offset, Imm cpol)
  __builtin_amdgcn_global_load_async_to_lds_b128(
      (__attribute__((address_space(1))) v4i*)(uintptr_t)g,
      (__attribute__((address_space(3))) v4i*)(unsigned int)(uintptr_t)l,
      0, 0);
#else
  *reinterpret_cast<uint4*>(l) = *reinterpret_cast<const uint4*>(g);
#endif
}

__device__ __forceinline__ void async_wait0() {
#if HAVE_ASYNC_LDS
#if defined(__has_builtin) && __has_builtin(__builtin_amdgcn_s_wait_asynccnt)
  __builtin_amdgcn_s_wait_asynccnt(0);
#else
  asm volatile("s_wait_asynccnt 0x0" ::: "memory");
#endif
#endif
}

// ---------------- WMMA tile loaders (wave32, gfx1250 layouts) ----------------
// A-matrix 16x32 f16 (activations, row-major [M,K]):
//   lane: row = base + (lane&15), h = lane>>4
//   VGPRs 0-3: K = kc + 8h + {0..7}; VGPRs 4-7: K = kc + 16 + 8h + {0..7}
__device__ __forceinline__ v16h load_a_tile(const f16* __restrict__ p) {
  union { v16h v; uint4 u[2]; } t;
  t.u[0] = *reinterpret_cast<const uint4*>(p);
  t.u[1] = *reinterpret_cast<const uint4*>(p + 16);
  return t.v;
}
// B-matrix 32x16 f16 from row-major [N,K] storage (B[k][n] = W[n][k]):
//   VGPRs 0-3: K = kc + 8h + {0..7}; VGPRs 4-7: K = kc + 16 + 8h + {0..7}
//   -> two 16-byte runs per lane (works for global or LDS pointers)
__device__ __forceinline__ v16h load_b_tile(const f16* p) {
  union { v16h v; uint4 u[2]; } t;
  t.u[0] = *reinterpret_cast<const uint4*>(p);
  t.u[1] = *reinterpret_cast<const uint4*>(p + 16);
  return t.v;
}

__device__ __forceinline__ float sig_(float x) { return 1.0f / (1.0f + __expf(-x)); }
__device__ __forceinline__ float tanh_(float x) {
  float e = __expf(-2.0f * fabsf(x));
  float t = (1.0f - e) / (1.0f + e);
  return copysignf(t, x);
}

#define WMMA_F16(A, Bv, C) \
  __builtin_amdgcn_wmma_f32_16x16x32_f16(false, (A), false, (Bv), (short)0, (C), false, false)

// LDS weight tile: 64 rows (gate*16 + n) x 64 f16 (one 64-wide K block),
// rows padded to 72 f16 (144B, 16B-aligned; stride 36 dwords -> conflict-free b128 reads)
#define LROW 72
#define LBUF (64 * LROW)

// Stage one 64x64 f16 weight block into LDS: 256 threads x 32B each.
__device__ __forceinline__ void stage_wblk(const f16* __restrict__ W, int ldK,
                                           int n0, int k0, f16* sbuf, int tid) {
  const int r   = tid >> 2;         // 0..63 : gate = r>>4, nrow = r&15
  const int seg = (tid & 3) << 4;   // 0,16,32,48 f16 elements (32B each)
  const f16* g = W + (size_t)((r >> 4) * HH + n0 + (r & 15)) * (size_t)ldK + k0 + seg;
  f16* l = sbuf + r * LROW + seg;
  stage16B(g, l);
  stage16B(g + 8, l + 8);
}

// One GEMM accumulation pass: acc[q] += A[m0..m0+15, 0..K) x W[q*HH+n0.., 0..K)^T
__device__ __forceinline__ void gemm_acc(const f16* arow,   // A + (m0+lr)*lda + 8*half
                                         const f16* __restrict__ W, int ldK, int K, int n0,
                                         int tid, int lr, int half, f16* sm,
                                         v8f& acc0, v8f& acc1, v8f& acc2, v8f& acc3) {
  stage_wblk(W, ldK, n0, 0, sm, tid);
  int buf = 0;
  for (int kb = 0; kb < K; kb += 64) {
    async_wait0();
    __syncthreads();                       // staged block visible to all waves
    if (kb + 64 < K) stage_wblk(W, ldK, n0, kb + 64, sm + (buf ^ 1) * LBUF, tid);
    const f16* sb = sm + buf * LBUF + (size_t)lr * LROW + 8 * half;
#pragma unroll
    for (int kc = 0; kc < 64; kc += 32) {
      // Batch the independent loads ahead of the WMMA group so the ds_loads
      // issue as one clause and drain with decremental waits.
      v16h a  = load_a_tile(arow + kb + kc);
      v16h b0 = load_b_tile(sb + kc + 0 * 16 * LROW);
      v16h b1 = load_b_tile(sb + kc + 1 * 16 * LROW);
      v16h b2 = load_b_tile(sb + kc + 2 * 16 * LROW);
      v16h b3 = load_b_tile(sb + kc + 3 * 16 * LROW);
      acc0 = WMMA_F16(a, b0, acc0);
      acc1 = WMMA_F16(a, b1, acc1);
      acc2 = WMMA_F16(a, b2, acc2);
      acc3 = WMMA_F16(a, b3, acc3);
    }
    buf ^= 1;
  }
  __syncthreads();                         // protect buffers before next pass restages
}

// ---------------- fused LSTM cell: G = x@Wih^T + h@Whh^T + bsum; gates ----------------
// grid.x = HH/16 = 64 (column tile within each gate), block = 256 (8 waves; wave -> 16 rows)
__global__ __launch_bounds__(256)
void lstm_cell_k(const f16* __restrict__ x, int ldx, int Kx,
                 const f16* __restrict__ hprev,
                 float* __restrict__ c,
                 const f16* __restrict__ Wih,   // [4H, Kx] row-major
                 const f16* __restrict__ Whh,   // [4H, HH] row-major
                 const float* __restrict__ bsum,// [4H] = bih + bhh
                 f16* __restrict__ hout)        // [BB, HH]
{
  __shared__ f16 smem[2 * LBUF];

  const int tid  = threadIdx.x;
  const int lane = tid & 31;
  const int wave = tid >> 5;
  const int lr   = lane & 15;
  const int half = lane >> 4;
  const int m0   = wave << 4;
  const int n0   = blockIdx.x << 4;
  const int n    = n0 + lr;

  v8f acc0 = {}, acc1 = {}, acc2 = {}, acc3 = {};  // gates i, f, g, o

  // ---- x @ Wih^T ----
  gemm_acc(x + (size_t)(m0 + lr) * (size_t)ldx + 8 * half,
           Wih, Kx, Kx, n0, tid, lr, half, smem, acc0, acc1, acc2, acc3);
  // ---- h @ Whh^T ----
  gemm_acc(hprev + (size_t)(m0 + lr) * HH + 8 * half,
           Whh, HH, HH, n0, tid, lr, half, smem, acc0, acc1, acc2, acc3);

  // All four gate values for a given (m, n) sit in the SAME lane/register slot.
  const float bi = bsum[0 * HH + n];
  const float bf = bsum[1 * HH + n];
  const float bg = bsum[2 * HH + n];
  const float bo = bsum[3 * HH + n];
#pragma unroll
  for (int r = 0; r < 8; ++r) {
    const int m = m0 + r + 8 * half;       // C/D layout: VGPR r -> M=r (lo half), M=8+r (hi half)
    const size_t idx = (size_t)m * HH + n;
    float gi = sig_(acc0[r] + bi);
    float gf = sig_(acc1[r] + bf);
    float gg = tanh_(acc2[r] + bg);
    float go = sig_(acc3[r] + bo);
    float cn = gf * c[idx] + gi * gg;
    c[idx] = cn;
    hout[idx] = (f16)(go * tanh_(cn));
  }
}

// ---------------- decoder head: out = h0 @ Wout^T + bout ----------------
// grid.x = FF/16 = 4, block = 256
__global__ __launch_bounds__(256)
void head_k(const f16* __restrict__ h0,      // [BB, HH]
            const f16* __restrict__ Wout,    // [FF, HH] row-major (f16)
            const float* __restrict__ bout,  // [FF]
            float* __restrict__ outp,        // d_out base, layout [BB, TT, FF]
            int tt,                          // time index to write
            f16* __restrict__ fb)            // [BB, FF] f16 feedback
{
  const int lane = threadIdx.x & 31;
  const int wave = threadIdx.x >> 5;
  const int lr   = lane & 15;
  const int half = lane >> 4;
  const int m0   = wave << 4;
  const int n    = (blockIdx.x << 4) + lr;

  v8f acc = {};
  const f16* arow = h0 + (size_t)(m0 + lr) * HH + 8 * half;
  const f16* brow = Wout + (size_t)n * HH + 8 * half;
  for (int kc = 0; kc < HH; kc += 32) {
    v16h a = load_a_tile(arow + kc);
    v16h b = load_b_tile(brow + kc);
    acc = WMMA_F16(a, b, acc);
  }
  const float bn = bout[n];
#pragma unroll
  for (int r = 0; r < 8; ++r) {
    const int m = m0 + r + 8 * half;
    float v = acc[r] + bn;
    outp[(size_t)m * TT * FF + (size_t)tt * FF + n] = v;
    fb[(size_t)m * FF + n] = (f16)v;
  }
}

// ---------------- prep kernels ----------------
__global__ void f32_to_f16_k(const float* __restrict__ s, f16* __restrict__ d, int nElem) {
  int i = blockIdx.x * blockDim.x + threadIdx.x;
  if (i < nElem) d[i] = (f16)s[i];
}
__global__ void bias_sum_k(const float* __restrict__ a, const float* __restrict__ b,
                           float* __restrict__ d, int nElem) {
  int i = blockIdx.x * blockDim.x + threadIdx.x;
  if (i < nElem) d[i] = a[i] + b[i];
}
__global__ void zero_f16_k(f16* __restrict__ d, int nElem) {
  int i = blockIdx.x * blockDim.x + threadIdx.x;
  if (i < nElem) d[i] = (f16)0.0f;
}
__global__ void zero_f32_k(float* __restrict__ d, int nElem) {
  int i = blockIdx.x * blockDim.x + threadIdx.x;
  if (i < nElem) d[i] = 0.0f;
}

// ---------------- host orchestration ----------------
extern "C" void kernel_launch(void* const* d_in, const int* in_sizes, int n_in,
                              void* d_out, int out_size, void* d_ws, size_t ws_size,
                              hipStream_t stream) {
  (void)in_sizes; (void)n_in; (void)out_size; (void)ws_size;

  const float* ts = (const float*)d_in[0];
  const float* W32[8];   // {Wih, Whh} x {enc0, enc1, dec0, dec1}
  const float* bih[4];
  const float* bhh[4];
  for (int k = 0; k < 4; ++k) {
    W32[2 * k + 0] = (const float*)d_in[1 + 4 * k + 0];
    W32[2 * k + 1] = (const float*)d_in[1 + 4 * k + 1];
    bih[k]         = (const float*)d_in[1 + 4 * k + 2];
    bhh[k]         = (const float*)d_in[1 + 4 * k + 3];
  }
  const float* Wout32 = (const float*)d_in[17];
  const float* bout   = (const float*)d_in[18];
  float* out = (float*)d_out;

  // ---- workspace carve-up (all slices 256B aligned) ----
  char* wsp = (char*)d_ws;
  auto carve = [&](size_t bytes) { char* p = wsp; wsp += (bytes + 255) & ~(size_t)255; return p; };
  const int wihK[4] = {FF, HH, FF, HH};

  f16* ts16 = (f16*)carve((size_t)BB * TT * FF * sizeof(f16));
  f16* W16[8];
  for (int k = 0; k < 4; ++k) {
    W16[2 * k + 0] = (f16*)carve((size_t)G4H * wihK[k] * sizeof(f16));
    W16[2 * k + 1] = (f16*)carve((size_t)G4H * HH * sizeof(f16));
  }
  f16*   Wout16 = (f16*)carve((size_t)FF * HH * sizeof(f16));
  float* bsum   = (float*)carve((size_t)4 * G4H * sizeof(float));
  f16*   hbuf   = (f16*)carve((size_t)2 * 2 * BB * HH * sizeof(f16)); // [layer][buf][B*H]
  float* cbuf   = (float*)carve((size_t)2 * BB * HH * sizeof(float)); // [layer][B*H]
  f16*   fb     = (f16*)carve((size_t)BB * FF * sizeof(f16));

  auto hp = [&](int l, int j) { return hbuf + ((size_t)l * 2 + j) * BB * HH; };
  auto conv = [&](const float* s, f16* d, int nElem) {
    f32_to_f16_k<<<(nElem + 255) / 256, 256, 0, stream>>>(s, d, nElem);
  };

  // ---- one-time prep (re-done every call; deterministic) ----
  conv(ts, ts16, BB * TT * FF);
  for (int k = 0; k < 4; ++k) {
    conv(W32[2 * k + 0], W16[2 * k + 0], G4H * wihK[k]);
    conv(W32[2 * k + 1], W16[2 * k + 1], G4H * HH);
    bias_sum_k<<<(G4H + 255) / 256, 256, 0, stream>>>(bih[k], bhh[k], bsum + (size_t)k * G4H, G4H);
  }
  conv(Wout32, Wout16, FF * HH);
  {
    int nh = 2 * 2 * BB * HH;
    zero_f16_k<<<(nh + 255) / 256, 256, 0, stream>>>(hbuf, nh);
    int nc = 2 * BB * HH;
    zero_f32_k<<<(nc + 255) / 256, 256, 0, stream>>>(cbuf, nc);
  }

  int cur0 = 0, cur1 = 0;

  // ---- encoder: t = 0 .. T-1 ----
  for (int t = 0; t < TT; ++t) {
    lstm_cell_k<<<HH / 16, 256, 0, stream>>>(
        ts16 + (size_t)t * FF, TT * FF, FF,
        hp(0, cur0), cbuf, W16[0], W16[1], bsum, hp(0, cur0 ^ 1));
    cur0 ^= 1;
    lstm_cell_k<<<HH / 16, 256, 0, stream>>>(
        hp(0, cur0), HH, HH,
        hp(1, cur1), cbuf + (size_t)BB * HH, W16[2], W16[3], bsum + G4H, hp(1, cur1 ^ 1));
    cur1 ^= 1;
  }

  // second output: c_enc[-1] (encoder layer-1 cell state) -> tail of d_out
  (void)hipMemcpyAsync(out + (size_t)BB * TT * FF, cbuf + (size_t)BB * HH,
                       (size_t)BB * HH * sizeof(float), hipMemcpyDeviceToDevice, stream);

  // ---- decoder: s = 0 .. T-1, writes time index T-1-s ----
  for (int s = 0; s < TT; ++s) {
    head_k<<<FF / 16, 256, 0, stream>>>(hp(0, cur0), Wout16, bout, out, TT - 1 - s, fb);
    lstm_cell_k<<<HH / 16, 256, 0, stream>>>(
        fb, FF, FF,
        hp(0, cur0), cbuf, W16[4], W16[5], bsum + 2 * G4H, hp(0, cur0 ^ 1));
    cur0 ^= 1;
    lstm_cell_k<<<HH / 16, 256, 0, stream>>>(
        hp(0, cur0), HH, HH,
        hp(1, cur1), cbuf + (size_t)BB * HH, W16[6], W16[7], bsum + 3 * G4H, hp(1, cur1 ^ 1));
    cur1 ^= 1;
  }
}
